// MLP_41755672051728
// MI455X (gfx1250) — matrix operand order repaired and checked by reading, not compiled, not run on hardware
//
#include <hip/hip_runtime.h>
#include <math.h>

typedef __attribute__((ext_vector_type(16))) _Float16 v16h;
typedef __attribute__((ext_vector_type(8)))  _Float16 v8h;
typedef __attribute__((ext_vector_type(8)))  float    v8f;

// ---------------- swizzled-weight layout in d_ws ----------------
// B-fragment (16x16x32 f16 WMMA, wave32): lane = (N%16) + 16*((K32>>3)&1),
// half h = (K32&7) + 8*(K32>=16).  One chunk = one (ks,nt) fragment = 32 lanes x 24 halves
// (16 used + 8 pad so each lane slot is 48B -> both v8h loads 16B-aligned).
#define LANE_STRIDE_W 24
#define W1_BASE 0    // W1: K=32  (KS=1),  N=256 (NT=16) -> chunks 0..15
#define W2_BASE 16   // W2: K=256 (KS=8),  N=128 (NT=8)  -> chunks 16..79
#define W3_BASE 80   // W3: K=128 (KS=4),  N=48  (NT=3)  -> chunks 80..91
#define TOTAL_W_ELEMS (8192 + 32768 + 6144)

__global__ void prep_weights(const float* __restrict__ W1,
                             const float* __restrict__ W2,
                             const float* __restrict__ W3,
                             _Float16* __restrict__ wsw) {
  int i = blockIdx.x * blockDim.x + threadIdx.x;
  int k, n, NT, base;
  float val;
  if (i < 8192) {                       // W1: 32 x 256, row-major (in,out)
    k = i >> 8; n = i & 255; NT = 16; base = W1_BASE;
    val = W1[i];
  } else if (i < 8192 + 32768) {        // W2: 256 x 128
    int j = i - 8192; k = j >> 7; n = j & 127; NT = 8; base = W2_BASE;
    val = W2[j];
  } else if (i < TOTAL_W_ELEMS) {       // W3: 128 x 34, zero-padded to 48
    int j = i - 40960; k = j / 48; n = j % 48; NT = 3; base = W3_BASE;
    val = (n < 34) ? W3[k * 34 + n] : 0.0f;
  } else {
    return;
  }
  int ks = k >> 5, kr = k & 31;
  int nt = n >> 4, nl = n & 15;
  int lane = nl + ((kr & 8) ? 16 : 0);
  int h = (kr & 7) + ((kr & 16) ? 8 : 0);
  wsw[(size_t)((base + ks * NT + nt) * 32 + lane) * LANE_STRIDE_W + h] = (_Float16)val;
}

// ---------------- fragment loaders ----------------
// A-matrix 16x32 f16 layout (ISA 7.12.2): lane holds M = lane%16;
// half h in [0,8): K = kb + h ; h in [8,16): K = kb + 8 + h ; kb = 8*(lane>=16).
// Row-major LDS with padded stride -> two aligned, conflict-free ds_load_b128.
__device__ __forceinline__ v16h load_afrag(const _Float16* base, int stride,
                                           int mrow0, int kbase, int lane) {
  int m = mrow0 + (lane & 15);
  int c = kbase + ((lane < 16) ? 0 : 8);
  const _Float16* p = base + m * stride + c;
  v8h lo = *(const v8h*)(p);        // K = c .. c+7
  v8h hi = *(const v8h*)(p + 16);   // K = c+16 .. c+23
  return __builtin_shufflevector(lo, hi, 0,1,2,3,4,5,6,7,8,9,10,11,12,13,14,15);
}

__device__ __forceinline__ v16h load_bfrag(const _Float16* __restrict__ wsw,
                                           int chunk, int lane) {
  const _Float16* p = wsw + (size_t)(chunk * 32 + lane) * LANE_STRIDE_W;
  v8h lo = *(const v8h*)(p);        // K%32 = 0..7  (+8 if lane>=16)
  v8h hi = *(const v8h*)(p + 8);    // K%32 = 16..23 (+8 if lane>=16)
  return __builtin_shufflevector(lo, hi, 0,1,2,3,4,5,6,7,8,9,10,11,12,13,14,15);
}

__device__ __forceinline__ v8f wmma_f16(v16h a, v16h b, v8f c) {
  return __builtin_amdgcn_wmma_f32_16x16x32_f16(false, a, false, b,
                                                (short)0, c, false, false);
}

// padded LDS strides (halves) -> 16B-aligned & bank-conflict-free b128 frag loads
#define X_STRIDE   40   // 32 cols
#define H1_STRIDE 264   // 256 cols
#define H2_STRIDE 136   // 128 cols
#define OUT_STRIDE 48   // floats, logits staging (reuses H1 space)

__launch_bounds__(128)
__global__ void mlp_kernel(const float* __restrict__ node,
                           const float* __restrict__ edge,
                           const float* __restrict__ b1,
                           const float* __restrict__ b2,
                           const float* __restrict__ b3,
                           const _Float16* __restrict__ wsw,
                           float* __restrict__ out, int Btot) {
  __shared__ __align__(16) _Float16 sX [4][32 * X_STRIDE];   // 10.0 KB
  __shared__ __align__(16) _Float16 sH1[4][32 * H1_STRIDE];  // 66.0 KB
  __shared__ __align__(16) _Float16 sH2[4][32 * H2_STRIDE];  // 34.0 KB

  const int lane  = threadIdx.x & 31;
  const int w     = threadIdx.x >> 5;
  const int ln    = lane & 15;
  const int sbase = (blockIdx.x * 4 + w) * 32;  // 32 samples per wave
  const int s     = sbase + lane;
  const bool valid = (s < Btot);

  _Float16* X  = sX[w];
  _Float16* H1 = sH1[w];
  _Float16* H2 = sH2[w];

  // ---- phase 0: per-lane compaction of first 4 nonzero rows ----
  float4 n0 = make_float4(0.f,0.f,0.f,0.f), n1 = n0, n2 = n0, n3 = n0;
  float4 e0 = n0, e1 = n0, e2 = n0, e3 = n0;
  if (valid) {
    const float4* pe = (const float4*)(edge + (size_t)s * 144);
    __builtin_prefetch(pe, 0, 1);
    const float4* pn = (const float4*)(node + (size_t)s * 76);
    int cnt = 0;
    for (int r = 0; r < 19; ++r) {
      float4 v = pn[r];
      bool nz = (v.x != 0.f) || (v.y != 0.f) || (v.z != 0.f) || (v.w != 0.f);
      if (nz && cnt < 4) {
        if (cnt == 0) n0 = v; else if (cnt == 1) n1 = v;
        else if (cnt == 2) n2 = v; else n3 = v;
        ++cnt;
      }
    }
    cnt = 0;
    for (int r = 0; r < 36; ++r) {
      float4 v = pe[r];
      bool nz = (v.x != 0.f) || (v.y != 0.f) || (v.z != 0.f) || (v.w != 0.f);
      if (nz && cnt < 4) {
        if (cnt == 0) e0 = v; else if (cnt == 1) e1 = v;
        else if (cnt == 2) e2 = v; else e3 = v;
        ++cnt;
      }
    }
  }
  {
    _Float16* xr = X + lane * X_STRIDE;
    v8h p0 = {(_Float16)n0.x,(_Float16)n0.y,(_Float16)n0.z,(_Float16)n0.w,
              (_Float16)n1.x,(_Float16)n1.y,(_Float16)n1.z,(_Float16)n1.w};
    v8h p1 = {(_Float16)n2.x,(_Float16)n2.y,(_Float16)n2.z,(_Float16)n2.w,
              (_Float16)n3.x,(_Float16)n3.y,(_Float16)n3.z,(_Float16)n3.w};
    v8h p2 = {(_Float16)e0.x,(_Float16)e0.y,(_Float16)e0.z,(_Float16)e0.w,
              (_Float16)e1.x,(_Float16)e1.y,(_Float16)e1.z,(_Float16)e1.w};
    v8h p3 = {(_Float16)e2.x,(_Float16)e2.y,(_Float16)e2.z,(_Float16)e2.w,
              (_Float16)e3.x,(_Float16)e3.y,(_Float16)e3.z,(_Float16)e3.w};
    *(v8h*)(xr +  0) = p0;  *(v8h*)(xr +  8) = p1;
    *(v8h*)(xr + 16) = p2;  *(v8h*)(xr + 24) = p3;
  }
  __syncthreads();

  // ---- layer 1: (32x32) x (32x256), relu ----
  {
    v16h a0 = load_afrag(X, X_STRIDE, 0,  0, lane);
    v16h a1 = load_afrag(X, X_STRIDE, 16, 0, lane);
    for (int nt = 0; nt < 16; ++nt) {
      v16h b = load_bfrag(wsw, W1_BASE + nt, lane);
      v8f c0 = {0.f,0.f,0.f,0.f,0.f,0.f,0.f,0.f};
      v8f c1 = c0;
      c0 = wmma_f16(a0, b, c0);
      c1 = wmma_f16(a1, b, c1);
      int   n    = nt * 16 + ln;
      float bias = b1[n];
      #pragma unroll
      for (int r = 0; r < 8; ++r) {
        int mrow = r + ((lane < 16) ? 0 : 8);        // C layout: M = r + 8*(lane>=16)
        float v0 = fmaxf(c0[r] + bias, 0.f);
        float v1 = fmaxf(c1[r] + bias, 0.f);
        H1[mrow * H1_STRIDE + n]        = (_Float16)v0;
        H1[(mrow + 16) * H1_STRIDE + n] = (_Float16)v1;
      }
    }
  }
  __syncthreads();

  // ---- layer 2: (32x256) x (256x128), relu ----
  {
    v16h a2[8][2];
    #pragma unroll
    for (int ks = 0; ks < 8; ++ks) {
      a2[ks][0] = load_afrag(H1, H1_STRIDE, 0,  ks * 32, lane);
      a2[ks][1] = load_afrag(H1, H1_STRIDE, 16, ks * 32, lane);
    }
    for (int nt = 0; nt < 8; ++nt) {
      v8f c0 = {0.f,0.f,0.f,0.f,0.f,0.f,0.f,0.f};
      v8f c1 = c0;
      #pragma unroll
      for (int ks = 0; ks < 8; ++ks) {
        v16h b = load_bfrag(wsw, W2_BASE + ks * 8 + nt, lane);
        c0 = wmma_f16(a2[ks][0], b, c0);
        c1 = wmma_f16(a2[ks][1], b, c1);
      }
      int   n    = nt * 16 + ln;
      float bias = b2[n];
      #pragma unroll
      for (int r = 0; r < 8; ++r) {
        int mrow = r + ((lane < 16) ? 0 : 8);
        float v0 = fmaxf(c0[r] + bias, 0.f);
        float v1 = fmaxf(c1[r] + bias, 0.f);
        H2[mrow * H2_STRIDE + n]        = (_Float16)v0;
        H2[(mrow + 16) * H2_STRIDE + n] = (_Float16)v1;
      }
    }
  }
  __syncthreads();

  // ---- layer 3: (32x128) x (128x48), 34 valid cols ----
  float* sOut = (float*)H1;  // 32 x 48 floats = 6 KB, fits in H1 region
  {
    v16h a3[4][2];
    #pragma unroll
    for (int ks = 0; ks < 4; ++ks) {
      a3[ks][0] = load_afrag(H2, H2_STRIDE, 0,  ks * 32, lane);
      a3[ks][1] = load_afrag(H2, H2_STRIDE, 16, ks * 32, lane);
    }
    __syncthreads();  // fence: H1 half-traffic done before float stores alias it
    for (int nt = 0; nt < 3; ++nt) {
      v8f c0 = {0.f,0.f,0.f,0.f,0.f,0.f,0.f,0.f};
      v8f c1 = c0;
      #pragma unroll
      for (int ks = 0; ks < 4; ++ks) {
        v16h b = load_bfrag(wsw, W3_BASE + ks * 3 + nt, lane);
        c0 = wmma_f16(a3[ks][0], b, c0);
        c1 = wmma_f16(a3[ks][1], b, c1);
      }
      int   n    = nt * 16 + ln;
      float bias = (n < 34) ? b3[n] : 0.f;
      #pragma unroll
      for (int r = 0; r < 8; ++r) {
        int mrow = r + ((lane < 16) ? 0 : 8);
        sOut[mrow * OUT_STRIDE + n]        = c0[r] + bias;
        sOut[(mrow + 16) * OUT_STRIDE + n] = c1[r] + bias;
      }
    }
  }
  __syncthreads();

  // ---- log-softmax: one sample per lane ----
  if (valid) {
    const float* xr = sOut + lane * OUT_STRIDE;
    float mx = xr[0];
    #pragma unroll
    for (int j = 1; j < 34; ++j) mx = fmaxf(mx, xr[j]);
    float sum = 0.f;
    #pragma unroll
    for (int j = 0; j < 34; ++j) sum += __expf(xr[j] - mx);
    float lse = mx + __logf(sum);
    float* o = out + (size_t)s * 34;
    #pragma unroll
    for (int j = 0; j < 34; ++j) o[j] = xr[j] - lse;
  }
}

extern "C" void kernel_launch(void* const* d_in, const int* in_sizes, int n_in,
                              void* d_out, int out_size, void* d_ws, size_t ws_size,
                              hipStream_t stream) {
  (void)n_in; (void)out_size; (void)ws_size;
  const float* node = (const float*)d_in[0];
  const float* edge = (const float*)d_in[1];
  const float* W1   = (const float*)d_in[2];
  const float* b1   = (const float*)d_in[3];
  const float* W2   = (const float*)d_in[4];
  const float* b2   = (const float*)d_in[5];
  const float* W3   = (const float*)d_in[6];
  const float* b3   = (const float*)d_in[7];
  float*    out = (float*)d_out;
  _Float16* wsw = (_Float16*)d_ws;   // needs 92*32*24*2 = 141312 bytes

  int Btot = in_sizes[0] / 76;

  prep_weights<<<(TOTAL_W_ELEMS + 255) / 256, 256, 0, stream>>>(W1, W2, W3, wsw);

  int blocks = (Btot + 127) / 128;   // 128 samples per block (4 waves x 32)
  mlp_kernel<<<blocks, 128, 0, stream>>>(node, edge, b1, b2, b3, wsw, out, Btot);
}
